// MQAttention_88785563942914
// MI455X (gfx1250) — matrix operand order repaired and checked by reading, not compiled, not run on hardware
//
#include <hip/hip_runtime.h>

typedef __attribute__((ext_vector_type(16))) __bf16 v16bf;
typedef __attribute__((ext_vector_type(8)))  __bf16 v8bf;
typedef __attribute__((ext_vector_type(2)))  __bf16 v2bf;
typedef __attribute__((ext_vector_type(8)))  float  v8f;

namespace {
constexpr int kDim   = 2048;
constexpr int kHeads = 16;
constexpr int kHd    = 128;
constexpr int kBatch = 8;
constexpr int kSeq   = 1024;
constexpr int kRows  = kBatch * kSeq;          // 8192
constexpr int kQkvN  = kDim + 2 * kHd;         // 2304
constexpr float kScale = 0.08838834764831845f; // 1/sqrt(128)
constexpr int kKpad = 136;                     // Ks row stride (bf16), 16B-aligned
constexpr int kVpad = 72;                      // Vt row stride (bf16), 16B-aligned
}

__device__ __forceinline__ __bf16 f2bf(float x) { return (__bf16)x; }

__device__ __forceinline__ v8f vzero8() {
  v8f z;
#pragma unroll
  for (int i = 0; i < 8; ++i) z[i] = 0.0f;
  return z;
}

// 32-bit LDS offset of a generic pointer to a __shared__ object.
template <typename T>
__device__ __forceinline__ unsigned lds_offset(T* p) {
  return (unsigned)(size_t)((__attribute__((address_space(3))) T*)p);
}

__device__ __forceinline__ v16bf cat8(v8bf a, v8bf b) {
  v16bf f;
#pragma unroll
  for (int e = 0; e < 8; ++e) { f[e] = a[e]; f[8 + e] = b[e]; }
  return f;
}

// A fragment (16x32 bf16, ISA 7.12.2): lane holds row M = lane&15.
// elems 0..7 -> K = k0 + (lane>>4)*8 + e ; elems 8..15 -> K = k0+16+(lane>>4)*8+e
__device__ __forceinline__ v16bf frag_a(const __bf16* __restrict__ p, int half) {
  return cat8(*(const v8bf*)(p + half * 8), *(const v8bf*)(p + 16 + half * 8));
}

// B fragment (32x16 bf16): lane holds column N = lane&15 of B (a row of B^T).
// elems 0..7 -> K = k0 + (lane>>4)*8 + e ; elems 8..15 -> K = k0+16+(lane>>4)*8+e
__device__ __forceinline__ v16bf frag_b(const __bf16* __restrict__ p, int half) {
  const __bf16* r = p + half * 8;
  return cat8(*(const v8bf*)r, *(const v8bf*)(r + 16));
}

// ---------------------------------------------------------------------------
// attention staging helpers
// ---------------------------------------------------------------------------
// Async global->LDS copy of a 64 x 128 bf16 key tile (1024 x 16B chunks).
__device__ __forceinline__ void stage_k_async(const __bf16* __restrict__ kb,
                                              __bf16 (*Ks)[kKpad], int tid) {
#pragma unroll
  for (int it = 0; it < 8; ++it) {
    const int c = tid + it * 128;
    const int s = c >> 4;
    const int d = (c & 15) * 8;
    const __bf16* gp = kb + (size_t)s * kQkvN + d;
    const unsigned loff = lds_offset(&Ks[s][d]);
    asm volatile("global_load_async_to_lds_b128 %0, %1, off"
                 :: "v"(loff), "v"(gp) : "memory");
  }
}

// V tile loads (held in VGPRs so the latency overlaps QK^T compute).
__device__ __forceinline__ void v_loads(const __bf16* __restrict__ vb, int tid,
                                        v8bf* r0, v8bf* r1) {
#pragma unroll
  for (int it = 0; it < 4; ++it) {
    const int c  = tid + it * 128;
    const int d0 = (c & 15) * 8;
    const int s  = (c >> 4) * 2;
    const __bf16* gp = vb + (size_t)s * kQkvN + d0;
    r0[it] = *(const v8bf*)gp;
    r1[it] = *(const v8bf*)(gp + kQkvN);
  }
}

// Transpose-store the held V tile into LDS: Vt[d][s] = V[s][d] (paired b32).
__device__ __forceinline__ void v_stores(__bf16 (*Vt)[kVpad], int tid,
                                         const v8bf* r0, const v8bf* r1) {
#pragma unroll
  for (int it = 0; it < 4; ++it) {
    const int c  = tid + it * 128;
    const int d0 = (c & 15) * 8;
    const int s  = (c >> 4) * 2;
#pragma unroll
    for (int e = 0; e < 8; ++e) {
      v2bf pr; pr[0] = r0[it][e]; pr[1] = r1[it][e];
      *(v2bf*)&Vt[d0 + e][s] = pr;
    }
  }
}

// ---------------------------------------------------------------------------
// f32 -> bf16 elementwise convert (n multiple of 8)
// ---------------------------------------------------------------------------
__global__ __launch_bounds__(256) void cvt_f32_bf16(const float* __restrict__ src,
                                                    __bf16* __restrict__ dst,
                                                    long long n) {
  const long long i = ((long long)blockIdx.x * blockDim.x + threadIdx.x) * 8;
  if (i + 8 > n) return;
  const float4 a = *(const float4*)(src + i);
  const float4 b = *(const float4*)(src + i + 4);
  v8bf o;
  o[0] = f2bf(a.x); o[1] = f2bf(a.y); o[2] = f2bf(a.z); o[3] = f2bf(a.w);
  o[4] = f2bf(b.x); o[5] = f2bf(b.y); o[6] = f2bf(b.z); o[7] = f2bf(b.w);
  *(v8bf*)(dst + i) = o;
}

// ---------------------------------------------------------------------------
// C = A @ W^T + bias.  A: MxK bf16 row-major, W: NxK bf16 row-major.
// 128 threads = 4 waves (2x2); each wave computes 32x64 (2x4 WMMA accums).
// Block tile 64x128.  OutT in {float, __bf16}.
// ---------------------------------------------------------------------------
template <typename OutT>
__global__ __launch_bounds__(128) void gemm_bias_wmma(
    const __bf16* __restrict__ A, const __bf16* __restrict__ W,
    const float* __restrict__ bias, OutT* __restrict__ C,
    int M, int N, int K)
{
  const int lane = threadIdx.x & 31;
  const int wave = threadIdx.x >> 5;
  const int half = lane >> 4;
  const int l15  = lane & 15;
  const int m0 = blockIdx.x * 64 + (wave >> 1) * 32;
  const int n0 = blockIdx.y * 128 + (wave & 1) * 64;

  v8f acc[2][4];
#pragma unroll
  for (int i = 0; i < 2; ++i)
#pragma unroll
    for (int j = 0; j < 4; ++j) acc[i][j] = vzero8();

  const __bf16* arow[2];
  const __bf16* wrow[4];
#pragma unroll
  for (int i = 0; i < 2; ++i) arow[i] = A + (size_t)(m0 + i * 16 + l15) * K;
#pragma unroll
  for (int j = 0; j < 4; ++j) wrow[j] = W + (size_t)(n0 + j * 16 + l15) * K;

  for (int k0 = 0; k0 < K; k0 += 32) {
#pragma unroll
    for (int i = 0; i < 2; ++i) __builtin_prefetch(arow[i] + k0 + 128, 0, 3);
#pragma unroll
    for (int j = 0; j < 4; ++j) __builtin_prefetch(wrow[j] + k0 + 128, 0, 3);

    v16bf af[2], bf[4];
#pragma unroll
    for (int i = 0; i < 2; ++i) af[i] = frag_a(arow[i] + k0, half);
#pragma unroll
    for (int j = 0; j < 4; ++j) bf[j] = frag_b(wrow[j] + k0, half);
#pragma unroll
    for (int i = 0; i < 2; ++i)
#pragma unroll
      for (int j = 0; j < 4; ++j)
        acc[i][j] = __builtin_amdgcn_wmma_f32_16x16x32_bf16(
            false, af[i], false, bf[j], (short)0, acc[i][j], false, false);
  }

#pragma unroll
  for (int i = 0; i < 2; ++i)
#pragma unroll
    for (int j = 0; j < 4; ++j)
#pragma unroll
      for (int v = 0; v < 8; ++v) {
        const int row = m0 + i * 16 + half * 8 + v;  // C layout: VGPR v + lane half
        const int col = n0 + j * 16 + l15;
        C[(size_t)row * N + col] = (OutT)(acc[i][j][v] + bias[col]);
      }
}

// ---------------------------------------------------------------------------
// Flash-attention (MQA, causal).  Grid: (T/64, H, B).  128 threads = 4 waves;
// wave w owns query rows [tq*64 + w*16, +16).  K/V shared across heads.
// Double-buffered LDS tiles: K staged via async-to-LDS (issued one iteration
// ahead, waited at iteration end), V prefetched to VGPRs then stored transposed.
// ---------------------------------------------------------------------------
__global__ __launch_bounds__(128) void mqa_flash_wmma(
    const __bf16* __restrict__ qkv, __bf16* __restrict__ attout)
{
  const int tq = blockIdx.x;
  const int h  = blockIdx.y;
  const int b  = blockIdx.z;
  const int tid  = threadIdx.x;
  const int lane = tid & 31;
  const int wave = tid >> 5;
  const int half = lane >> 4;
  const int l15  = lane & 15;

  __shared__ __bf16 Ks[2][64][kKpad];   // key tiles, row-major (s, dk)
  __shared__ __bf16 Vt[2][kHd][kVpad];  // value tiles, transposed (d, s)
  __shared__ float  S[4][16][64];
  __shared__ __bf16 P[4][16][64];
  __shared__ float  Mrow[4][16];
  __shared__ float  Lrow[4][16];
  __shared__ float  Alpha[4][16];

  const size_t seq0 = (size_t)b * kSeq;
  const int qrow0 = tq * 64 + wave * 16;

  if (lane < 16) { Mrow[wave][lane] = -__builtin_inff(); Lrow[wave][lane] = 0.0f; }

  v8f o[8];
#pragma unroll
  for (int d = 0; d < 8; ++d) o[d] = vzero8();

  // Preload Q fragments for this wave's 16 rows (K = 128 = 4 chunks of 32).
  v16bf qf[4];
  const __bf16* qrow = qkv + (seq0 + qrow0 + l15) * kQkvN + (size_t)h * kHd;
#pragma unroll
  for (int kc = 0; kc < 4; ++kc) qf[kc] = frag_a(qrow + kc * 32, half);

  // ---- prologue: stage tiles for j = 0 into buffer 0 ----
  {
    const __bf16* kb0 = qkv + seq0 * kQkvN + kDim;
    stage_k_async(kb0, Ks[0], tid);
    v8bf r0[4], r1[4];
    v_loads(kb0 + kHd, tid, r0, r1);
    v_stores(Vt[0], tid, r0, r1);
    asm volatile("s_wait_asynccnt 0" ::: "memory");
  }
  __syncthreads();

  for (int j = 0; j <= tq; ++j) {
    const int buf = j & 1;
    const __bf16* kbn = qkv + (seq0 + (size_t)(j + 1) * 64) * kQkvN + kDim;

    // issue next iteration's staging early: async K copy + V loads to VGPRs
    v8bf vr0[4], vr1[4];
    if (j < tq) {
      stage_k_async(kbn, Ks[buf ^ 1], tid);
      v_loads(kbn + kHd, tid, vr0, vr1);
    }

    // ---- S strip = Q @ K^T from LDS (buffer j) ----
#pragma unroll
    for (int nt = 0; nt < 4; ++nt) {
      v8f s = vzero8();
      const __bf16* krow = &Ks[buf][nt * 16 + l15][0];
#pragma unroll
      for (int kc = 0; kc < 4; ++kc) {
        v16bf kf = frag_b(krow + kc * 32, half);
        s = __builtin_amdgcn_wmma_f32_16x16x32_bf16(false, qf[kc], false, kf,
                                                    (short)0, s, false, false);
      }
#pragma unroll
      for (int v = 0; v < 8; ++v) S[wave][half * 8 + v][nt * 16 + l15] = s[v];
    }

    // complete next V tile: transpose-store into spare buffer
    if (j < tq) v_stores(Vt[buf ^ 1], tid, vr0, vr1);
    __syncthreads();

    // ---- online softmax: lane pair (l15, l15+16) owns query row qrow0+l15 ----
    {
      const int r = l15;
      const int c0 = half * 32;
      const int grow = qrow0 + r;
      const float mold = Mrow[wave][r];
      float mx = mold;
#pragma unroll 8
      for (int c = 0; c < 32; ++c) {
        const int gc = j * 64 + c0 + c;
        const float sv = (gc <= grow) ? S[wave][r][c0 + c] * kScale
                                      : -__builtin_inff();
        S[wave][r][c0 + c] = sv;
        mx = fmaxf(mx, sv);
      }
      mx = fmaxf(mx, __shfl_xor(mx, 16));
      const float al = __expf(mold - mx);
      float rs = 0.0f;
#pragma unroll 8
      for (int c = 0; c < 32; ++c) {
        const float p = __expf(S[wave][r][c0 + c] - mx);
        P[wave][r][c0 + c] = f2bf(p);
        rs += p;
      }
      rs += __shfl_xor(rs, 16);
      if (half == 0) {
        Mrow[wave][r]  = mx;
        Lrow[wave][r]  = Lrow[wave][r] * al + rs;
        Alpha[wave][r] = al;
      }
    }
    __syncthreads();

    // ---- rescale O by alpha, then O += P @ V (fragments from LDS) ----
    float asc[8];
#pragma unroll
    for (int v = 0; v < 8; ++v) asc[v] = Alpha[wave][half * 8 + v];
#pragma unroll
    for (int d = 0; d < 8; ++d)
#pragma unroll
      for (int v = 0; v < 8; ++v) o[d][v] *= asc[v];

    v16bf pf[2];
#pragma unroll
    for (int kc = 0; kc < 2; ++kc)
      pf[kc] = frag_a(&P[wave][l15][kc * 32], half);

#pragma unroll
    for (int dt = 0; dt < 8; ++dt) {
      const __bf16* vrow = &Vt[buf][dt * 16 + l15][0];
#pragma unroll
      for (int kc = 0; kc < 2; ++kc) {
        v16bf vf = frag_b(vrow + kc * 32, half);
        o[dt] = __builtin_amdgcn_wmma_f32_16x16x32_bf16(false, pf[kc], false, vf,
                                                        (short)0, o[dt], false, false);
      }
    }

    // next iteration's async K tile must be resident before the barrier
    asm volatile("s_wait_asynccnt 0" ::: "memory");
    __syncthreads();
  }

  // ---- normalize by row sum and store bf16 at (b, t, h*128 + d) ----
  float inv[8];
#pragma unroll
  for (int v = 0; v < 8; ++v) inv[v] = 1.0f / Lrow[wave][half * 8 + v];
#pragma unroll
  for (int dt = 0; dt < 8; ++dt)
#pragma unroll
    for (int v = 0; v < 8; ++v) {
      const int r = half * 8 + v;
      const size_t row = seq0 + (size_t)(tq * 64 + wave * 16 + r);
      attout[row * kDim + (size_t)h * kHd + dt * 16 + l15] = f2bf(o[dt][v] * inv[v]);
    }
}

// ---------------------------------------------------------------------------

extern "C" void kernel_launch(void* const* d_in, const int* in_sizes, int n_in,
                              void* d_out, int out_size, void* d_ws, size_t ws_size,
                              hipStream_t stream) {
  (void)in_sizes; (void)n_in; (void)out_size; (void)ws_size;
  const float* x    = (const float*)d_in[0];   // (8,1024,2048)
  const float* Wqkv = (const float*)d_in[1];   // (2304,2048)
  const float* bqkv = (const float*)d_in[2];   // (2304,)
  const float* Wout = (const float*)d_in[3];   // (2048,2048)
  const float* bout = (const float*)d_in[4];   // (2048,)
  float* out = (float*)d_out;                  // (8,1024,2048)

  // bf16 workspace layout
  __bf16* xbf    = (__bf16*)d_ws;                                  // 8192*2048
  __bf16* wqkvbf = xbf    + (size_t)kRows * kDim;                  // 2304*2048
  __bf16* woutbf = wqkvbf + (size_t)kQkvN * kDim;                  // 2048*2048
  __bf16* qkvbf  = woutbf + (size_t)kDim * kDim;                   // 8192*2304
  __bf16* attbf  = qkvbf  + (size_t)kRows * kQkvN;                 // 8192*2048

  const long long nx  = (long long)kRows * kDim;
  const long long nwq = (long long)kQkvN * kDim;
  const long long nwo = (long long)kDim * kDim;
  dim3 cblk(256);
  cvt_f32_bf16<<<dim3((unsigned)(nx  / (8 * 256))), cblk, 0, stream>>>(x, xbf, nx);
  cvt_f32_bf16<<<dim3((unsigned)(nwq / (8 * 256))), cblk, 0, stream>>>(Wqkv, wqkvbf, nwq);
  cvt_f32_bf16<<<dim3((unsigned)(nwo / (8 * 256))), cblk, 0, stream>>>(Wout, woutbf, nwo);

  dim3 blk(128);
  gemm_bias_wmma<__bf16><<<dim3(kRows / 64, kQkvN / 128), blk, 0, stream>>>(
      xbf, wqkvbf, bqkv, qkvbf, kRows, kQkvN, kDim);
  mqa_flash_wmma<<<dim3(kSeq / 64, kHeads, kBatch), blk, 0, stream>>>(qkvbf, attbf);
  gemm_bias_wmma<float><<<dim3(kRows / 64, kDim / 128), blk, 0, stream>>>(
      attbf, woutbf, bout, out, kRows, kDim, kDim);
}
